// Speller_32993938768082
// MI455X (gfx1250) — matrix operand order repaired and checked by reading, not compiled, not run on hardware
//
#include <hip/hip_runtime.h>

// ---------------------------------------------------------------------------
// Speller (3-layer LSTM + dot attention + tied-embedding output) for gfx1250.
// bf16 WMMA GEMMs (fp32 accumulate), fp32 cell state / softmax.
// B=128, T=400, TE=1600, V=31, EMB=256, H=512, P=256.
// ---------------------------------------------------------------------------

typedef unsigned short ushort_t;
typedef __attribute__((ext_vector_type(16))) __bf16 v16bf;
typedef __attribute__((ext_vector_type(8)))  float  v8f;

#if __has_builtin(__builtin_amdgcn_sched_barrier)
#define SCHED_BARRIER() __builtin_amdgcn_sched_barrier(0)
#else
#define SCHED_BARRIER()
#endif

#define Bb   128
#define Tt   400
#define TE   1600
#define Vv   31
#define EMB  256
#define Hh   512
#define Pp   256
#define SOS_TOK 1

__device__ __forceinline__ float bf2f(ushort_t v) {
    return __uint_as_float(((unsigned)v) << 16);
}
__device__ __forceinline__ float bflo(unsigned u) {     // low bf16 of packed dword
    return __uint_as_float(u << 16);
}
__device__ __forceinline__ float bfhi(unsigned u) {     // high bf16 of packed dword
    return __uint_as_float(u & 0xffff0000u);
}
__device__ __forceinline__ ushort_t f2bf(float x) {
    unsigned u = __float_as_uint(x);
    unsigned r = u + 0x7FFFu + ((u >> 16) & 1u);        // round-to-nearest-even
    return (ushort_t)(r >> 16);
}
__device__ __forceinline__ float sigm(float x) {
    return 1.0f / (1.0f + __expf(-x));
}

// --------------------------- one-time conversion ---------------------------

__global__ void cvt_bf16_kernel(const float* __restrict__ s, ushort_t* __restrict__ d, size_t n) {
    size_t i = (size_t)blockIdx.x * blockDim.x + threadIdx.x;
    if (i < n) d[i] = f2bf(s[i]);
}

// Convert W (Nvalid x Ktot fp32, optionally concat of W1[N x K1] and W2[N x (Ktot-K1)])
// into the WMMA B-operand fragment layout:
//   dst[(((n_tile*ntk + k_tile)*32 + lane)*16 + e] = W[n][k]
//   n = n_tile*16 + (lane&15),  k = k_tile*32 + (lane>=16?8:0) + (e>=8?16:0) + (e&7)
__global__ void swz_convert_kernel(const float* __restrict__ W1, const float* __restrict__ W2,
                                   ushort_t* __restrict__ dst,
                                   int Npad, int Nvalid, int K1, int Ktot) {
    size_t idx = (size_t)blockIdx.x * blockDim.x + threadIdx.x;
    size_t total = (size_t)Npad * (size_t)Ktot;
    if (idx >= total) return;
    int e    = (int)(idx & 15);
    int lane = (int)((idx >> 4) & 31);
    size_t tk = idx >> 9;
    int ntk = Ktot >> 5;
    int k_tile = (int)(tk % (size_t)ntk);
    int n_tile = (int)(tk / (size_t)ntk);
    int n = n_tile * 16 + (lane & 15);
    int k = k_tile * 32 + ((lane >= 16) ? 8 : 0) + ((e >= 8) ? 16 : 0) + (e & 7);
    float v = 0.0f;
    if (n < Nvalid) {
        if (k < K1) v = W1[(size_t)n * K1 + k];
        else        v = W2[(size_t)n * (Ktot - K1) + (k - K1)];
    }
    dst[idx] = f2bf(v);
}

__global__ void bias_kernel(const float* __restrict__ b1, const float* __restrict__ b2,
                            float* __restrict__ dst, int npad, int nvalid) {
    int n = blockIdx.x * blockDim.x + threadIdx.x;
    if (n >= npad) return;
    float v = 0.0f;
    if (n < nvalid) v = b1[n] + (b2 ? b2[n] : 0.0f);
    dst[n] = v;
}

__global__ void fill0_kernel(unsigned* __restrict__ p, size_t ndw) {
    size_t i = (size_t)blockIdx.x * blockDim.x + threadIdx.x;
    if (i < ndw) p[i] = 0u;
}

// ------------------------------- WMMA GEMM ---------------------------------
// D(M x Npad) = A(M x K, bf16 row-major, row stride lda) * Bsw + bias.
// 4 waves/block; each wave computes NT adjacent 16x16 tiles of one m-row-block.
// K is compile-time -> all loads are immediate-offset b128 from two bases.
// Explicit 2-stage software pipeline, ENFORCED with sched_barrier(0) so the
// machine scheduler cannot sink the prefetch loads back to their consumers:
//   loads(kt+1) | barrier | wmma(kt) | barrier | loads(kt+2) | ...
// -> stage kt+1 fragments stay live across the stage-kt WMMA group (distinct
// registers) and waits become partial (stage-kt loads only), hiding L2 latency
// inside a single wave (critical: this GEMM runs at ~1 wave/SIMD occupancy).
// MODE 0: fp32 out (row stride ldc).  MODE 1: bf16(tanh(x)) out.
union AFrag { struct { uint4 lo; uint4 hi; } u; v16bf v; };

template <int MODE, int NT, int K>
__global__ __launch_bounds__(128)
void gemm_bf16_wmma(const ushort_t* __restrict__ A, int lda,
                    const ushort_t* __restrict__ Bsw,
                    const float* __restrict__ bias,
                    float* __restrict__ outf, ushort_t* __restrict__ outb,
                    int ldc, int Npad, int nvalid) {
    constexpr int NTK = K >> 5;                         // k-tiles
    constexpr int TILE_STRIDE = NTK * 512;              // elements between n-tile streams
    const int tid  = threadIdx.x;
    const int lane = tid & 31;
    const int wave = tid >> 5;
    const int nt0  = (blockIdx.x * 4 + wave) * NT;      // first n-tile of this wave
    if (nt0 * 16 >= Npad) return;                       // wave-uniform: EXEC stays all-ones
    const int m0  = blockIdx.y * 16;
    const int lhalf = lane >> 4;                        // 0 or 1

    v8f acc[NT];
#pragma unroll
    for (int j = 0; j < NT; ++j) {
        float bv = bias[(nt0 + j) * 16 + (lane & 15)];  // bias depends only on N
#pragma unroll
        for (int r = 0; r < 8; ++r) acc[j][r] = bv;
    }

    const ushort_t* arow  = A + (size_t)(m0 + (lane & 15)) * lda + lhalf * 8;
    const ushort_t* bbase = Bsw + ((size_t)nt0 * NTK * 32 + lane) * 16;

    struct Stage { AFrag af; v16bf bf[NT]; };
    Stage st[2];

    auto loadStage = [&](Stage& s, int kt) {
        const ushort_t* ar = arow + kt * 32;
        s.af.u.lo = *(const uint4*)(ar);                // k in [kb, kb+8)
        s.af.u.hi = *(const uint4*)(ar + 16);           // k in [kb+16, kb+24)
        const ushort_t* bp = bbase + kt * 512;
#pragma unroll
        for (int j = 0; j < NT; ++j)                    // immediate-offset b128 loads
            s.bf[j] = *(const v16bf*)(bp + (size_t)j * TILE_STRIDE);
    };
    auto mmaStage = [&](Stage& s) {
#pragma unroll
        for (int j = 0; j < NT; ++j)
            acc[j] = __builtin_amdgcn_wmma_f32_16x16x32_bf16(
                         false, s.af.v, false, s.bf[j], (short)0, acc[j], false, false);
    };

    loadStage(st[0], 0);
    SCHED_BARRIER();
#pragma unroll
    for (int kt = 0; kt < NTK - 1; ++kt) {
        loadStage(st[(kt + 1) & 1], kt + 1);            // prefetch next stage
        SCHED_BARRIER();                                // loads must stay ABOVE the WMMAs
        mmaStage(st[kt & 1]);                           // consume current stage
        SCHED_BARRIER();                                // WMMAs must stay ABOVE next loads
    }
    mmaStage(st[(NTK - 1) & 1]);

    const int mB = m0 + (lhalf << 3);
#pragma unroll
    for (int j = 0; j < NT; ++j) {
        int n = (nt0 + j) * 16 + (lane & 15);
        if (n < nvalid) {
#pragma unroll
            for (int r = 0; r < 8; ++r) {
                size_t o = (size_t)(mB + r) * ldc + n;
                if (MODE == 0) outf[o] = acc[j][r];
                else           outb[o] = f2bf(tanhf(acc[j][r]));
            }
        }
    }
}

// ----------------------------- per-step kernels ----------------------------

// token embedding gather into xin1[:, 0:EMB] (bf16)
__global__ void emb_kernel(const int* __restrict__ y, const ushort_t* __restrict__ Eemb,
                           ushort_t* __restrict__ xin1, int t) {
    int idx = blockIdx.x * blockDim.x + threadIdx.x;    // 128*256
    int b = idx >> 8, e = idx & 255;
    int tok = (t == 0) ? SOS_TOK : y[b * Tt + (t - 1)];
    xin1[(size_t)b * 1024 + e] = Eemb[tok * EMB + e];
}

// LSTM gate nonlinearity: gates g (B x 4H fp32, order i,f,g,o), c state fp32,
// write h as bf16 into two consumer slots (pre-offset base, given row stride).
__global__ void lstm_pw_kernel(const float* __restrict__ g, float* __restrict__ c,
                               ushort_t* __restrict__ hd0, int ld0,
                               ushort_t* __restrict__ hd1, int ld1) {
    int idx = blockIdx.x * blockDim.x + threadIdx.x;    // 128*512
    int b = idx >> 9, u = idx & 511;
    const float* gr = g + (size_t)b * 2048;
    float gi = sigm(gr[u]);
    float gf = sigm(gr[512 + u]);
    float gg = tanhf(gr[1024 + u]);
    float go = sigm(gr[1536 + u]);
    float cc = gf * c[idx] + gi * gg;
    c[idx] = cc;
    ushort_t hb = f2bf(go * tanhf(cc));
    hd0[(size_t)b * ld0 + u] = hb;
    hd1[(size_t)b * ld1 + u] = hb;
}

// Dot attention: one block per batch row. energy->softmax in LDS, weights to
// attention_plot output, context bf16 into next-step xin1 and the CDN concat.
// keys read as uint4 (8 bf16 / b128 load); values read as packed dwords with
// full-row coalescing (128 threads x u32 = 512B row), two t-halves combined in LDS.
__global__ void attn_kernel(const float* __restrict__ qbuf,
                            const ushort_t* __restrict__ keysb,
                            const ushort_t* __restrict__ valsb,
                            float* __restrict__ attn_out, int tstep,
                            ushort_t* __restrict__ xin1, ushort_t* __restrict__ cdn) {
    __shared__ float qs[Pp];
    __shared__ float en[TE];            // energies -> weights -> (reused) ctx partials
    __shared__ float red[256];
    const int b = blockIdx.x, tid = threadIdx.x;

    qs[tid] = qbuf[(size_t)b * Pp + tid];
    __syncthreads();

    // energy[t] = q . keys[b,t,:]
    for (int t = tid; t < TE; t += 256) {
        const uint4* kr4 = (const uint4*)(keysb + ((size_t)b * TE + t) * Pp);
        float acc = 0.0f;
#pragma unroll 8
        for (int j = 0; j < 32; ++j) {
            uint4 kv = kr4[j];
            const float* qp = qs + j * 8;
            acc = fmaf(qp[0], bflo(kv.x), acc);
            acc = fmaf(qp[1], bfhi(kv.x), acc);
            acc = fmaf(qp[2], bflo(kv.y), acc);
            acc = fmaf(qp[3], bfhi(kv.y), acc);
            acc = fmaf(qp[4], bflo(kv.z), acc);
            acc = fmaf(qp[5], bfhi(kv.z), acc);
            acc = fmaf(qp[6], bflo(kv.w), acc);
            acc = fmaf(qp[7], bfhi(kv.w), acc);
        }
        en[t] = acc;
    }
    __syncthreads();

    // softmax over TE
    float m = -3.0e38f;
    for (int t = tid; t < TE; t += 256) m = fmaxf(m, en[t]);
    red[tid] = m; __syncthreads();
    for (int s = 128; s > 0; s >>= 1) {
        if (tid < s) red[tid] = fmaxf(red[tid], red[tid + s]);
        __syncthreads();
    }
    m = red[0];
    __syncthreads();

    float sum = 0.0f;
    for (int t = tid; t < TE; t += 256) { float e = __expf(en[t] - m); en[t] = e; sum += e; }
    red[tid] = sum; __syncthreads();
    for (int s = 128; s > 0; s >>= 1) {
        if (tid < s) red[tid] += red[tid + s];
        __syncthreads();
    }
    float inv = 1.0f / red[0];
    __syncthreads();

    for (int t = tid; t < TE; t += 256) {
        float w = en[t] * inv;
        en[t] = w;
        attn_out[((size_t)b * Tt + tstep) * TE + t] = w;
    }
    __syncthreads();

    // ctx[p] = sum_t w[t] * values[b,t,p]
    // thread (half, i): columns {2i, 2i+1}, rows t = half, half+2, ...
    const int half = tid >> 7;
    const int p0   = (tid & 127) * 2;
    const ushort_t* vb = valsb + (size_t)b * TE * Pp;
    float a0 = 0.0f, a1 = 0.0f;
    for (int t = half; t < TE; t += 2) {
        unsigned w2 = *(const unsigned*)(vb + (size_t)t * Pp + p0);
        float wt = en[t];
        a0 = fmaf(wt, bflo(w2), a0);
        a1 = fmaf(wt, bfhi(w2), a1);
    }
    __syncthreads();                    // all reads of en done
    en[half * 256 + p0]     = a0;       // reuse en[0..511] as ctx partials
    en[half * 256 + p0 + 1] = a1;
    __syncthreads();
    {
        float v = en[tid] + en[256 + tid];
        ushort_t cb = f2bf(v);
        xin1[(size_t)b * 1024 + Pp + tid] = cb;   // next-step x = [emb | ctx | h1]
        cdn [(size_t)b * 768  + Hh + tid] = cb;   // cdn = [h3 | ctx]
    }
}

// ------------------------------- host driver -------------------------------

extern "C" void kernel_launch(void* const* d_in, const int* in_sizes, int n_in,
                              void* d_out, int out_size, void* d_ws, size_t ws_size,
                              hipStream_t stream) {
    const int*   y      = (const int*)  d_in[0];
    const float* keys   = (const float*)d_in[1];
    const float* values = (const float*)d_in[2];
    const float* E      = (const float*)d_in[3];
    const float* Wih1 = (const float*)d_in[4],  *bih1 = (const float*)d_in[5];
    const float* Whh1 = (const float*)d_in[6],  *bhh1 = (const float*)d_in[7];
    const float* Wih2 = (const float*)d_in[8],  *bih2 = (const float*)d_in[9];
    const float* Whh2 = (const float*)d_in[10], *bhh2 = (const float*)d_in[11];
    const float* Wih3 = (const float*)d_in[12], *bih3 = (const float*)d_in[13];
    const float* Whh3 = (const float*)d_in[14], *bhh3 = (const float*)d_in[15];
    const float* Wq   = (const float*)d_in[16], *bq   = (const float*)d_in[17];
    const float* Wc   = (const float*)d_in[18], *bc   = (const float*)d_in[19];
    const float* bprob = (const float*)d_in[20];

    float* raw_out  = (float*)d_out;                                 // (B,T,V)
    float* attn_out = raw_out + (size_t)Bb * Tt * Vv;                // (B,T,TE)

    // ---- workspace bump allocator (256B aligned) ----
    unsigned char* ws = (unsigned char*)d_ws;
    size_t off = 0;
    auto alloc = [&](size_t bytes) -> void* {
        void* p = ws + off;
        off = (off + bytes + 255) & ~(size_t)255;
        return p;
    };
    ushort_t* keysb = (ushort_t*)alloc((size_t)Bb * TE * Pp * 2);
    ushort_t* valsb = (ushort_t*)alloc((size_t)Bb * TE * Pp * 2);
    ushort_t* W1sw  = (ushort_t*)alloc((size_t)2048 * 1024 * 2);
    ushort_t* W2sw  = (ushort_t*)alloc((size_t)2048 * 1024 * 2);
    ushort_t* W3sw  = (ushort_t*)alloc((size_t)2048 * 1024 * 2);
    ushort_t* Wqsw  = (ushort_t*)alloc((size_t)256 * 512 * 2);
    ushort_t* Wcsw  = (ushort_t*)alloc((size_t)256 * 768 * 2);
    ushort_t* Epsw  = (ushort_t*)alloc((size_t)32 * 256 * 2);        // tied E^T, N padded to 32
    ushort_t* Eemb  = (ushort_t*)alloc((size_t)32 * 256 * 2);        // plain bf16 E (gather)
    float*    b1s   = (float*)alloc(2048 * 4);
    float*    b2s   = (float*)alloc(2048 * 4);
    float*    b3s   = (float*)alloc(2048 * 4);
    float*    bqp   = (float*)alloc(256 * 4);
    float*    bcp   = (float*)alloc(256 * 4);
    float*    bpp   = (float*)alloc(32 * 4);
    ushort_t* xin1  = (ushort_t*)alloc((size_t)Bb * 1024 * 2);       // [emb|ctx|h1]
    ushort_t* xin2  = (ushort_t*)alloc((size_t)Bb * 1024 * 2);       // [h1|h2]
    ushort_t* xin3  = (ushort_t*)alloc((size_t)Bb * 1024 * 2);       // [h2|h3]
    ushort_t* cdn   = (ushort_t*)alloc((size_t)Bb * 768 * 2);        // [h3|ctx]
    ushort_t* sbuf  = (ushort_t*)alloc((size_t)Bb * 256 * 2);        // tanh(cdn@Wc^T+bc)
    float*    c1    = (float*)alloc((size_t)Bb * Hh * 4);
    float*    c2    = (float*)alloc((size_t)Bb * Hh * 4);
    float*    c3    = (float*)alloc((size_t)Bb * Hh * 4);
    float*    gates = (float*)alloc((size_t)Bb * 2048 * 4);
    float*    qbuf  = (float*)alloc((size_t)Bb * Pp * 4);
    (void)in_sizes; (void)n_in; (void)out_size; (void)ws_size;

    // ---- one-time conversions (re-run each call: deterministic) ----
    {
        size_t n = (size_t)Bb * TE * Pp;
        int blk = (int)((n + 255) / 256);
        cvt_bf16_kernel<<<blk, 256, 0, stream>>>(keys,   keysb, n);
        cvt_bf16_kernel<<<blk, 256, 0, stream>>>(values, valsb, n);
    }
    cvt_bf16_kernel<<<(31 * 256 + 255) / 256, 256, 0, stream>>>(E, Eemb, (size_t)31 * 256);

    swz_convert_kernel<<<(2048 * 1024) / 256, 256, 0, stream>>>(Wih1, Whh1, W1sw, 2048, 2048, 512, 1024);
    swz_convert_kernel<<<(2048 * 1024) / 256, 256, 0, stream>>>(Wih2, Whh2, W2sw, 2048, 2048, 512, 1024);
    swz_convert_kernel<<<(2048 * 1024) / 256, 256, 0, stream>>>(Wih3, Whh3, W3sw, 2048, 2048, 512, 1024);
    swz_convert_kernel<<<(256 * 512)   / 256, 256, 0, stream>>>(Wq, Wq, Wqsw, 256, 256, 512, 512);
    swz_convert_kernel<<<(256 * 768)   / 256, 256, 0, stream>>>(Wc, Wc, Wcsw, 256, 256, 768, 768);
    swz_convert_kernel<<<(32 * 256)    / 256, 256, 0, stream>>>(E,  E,  Epsw, 32, 31, 256, 256);

    bias_kernel<<<8, 256, 0, stream>>>(bih1, bhh1, b1s, 2048, 2048);
    bias_kernel<<<8, 256, 0, stream>>>(bih2, bhh2, b2s, 2048, 2048);
    bias_kernel<<<8, 256, 0, stream>>>(bih3, bhh3, b3s, 2048, 2048);
    bias_kernel<<<1, 256, 0, stream>>>(bq, nullptr, bqp, 256, 256);
    bias_kernel<<<1, 256, 0, stream>>>(bc, nullptr, bcp, 256, 256);
    bias_kernel<<<1, 256, 0, stream>>>(bprob, nullptr, bpp, 32, 31);

    // zero initial state: c1..c3 (fp32), xin1..3 + cdn (bf16)
    auto zero = [&](void* p, size_t bytes) {
        size_t ndw = bytes / 4;
        fill0_kernel<<<(int)((ndw + 255) / 256), 256, 0, stream>>>((unsigned*)p, ndw);
    };
    zero(c1, (size_t)Bb * Hh * 4);  zero(c2, (size_t)Bb * Hh * 4);  zero(c3, (size_t)Bb * Hh * 4);
    zero(xin1, (size_t)Bb * 1024 * 2); zero(xin2, (size_t)Bb * 1024 * 2);
    zero(xin3, (size_t)Bb * 1024 * 2); zero(cdn,  (size_t)Bb * 768 * 2);

    // ---- sequential decode ----
    // grid.x = Npad / (4 waves * NT tiles * 16)
    const dim3 gLSTM(2048 / 256, Bb / 16);  // NT=4
    const dim3 g256(1, Bb / 16);            // NT=4, Npad=256
    const dim3 gV(1, Bb / 16);              // NT=1, Npad=32

    for (int t = 0; t < Tt; ++t) {
        emb_kernel<<<Bb, 256, 0, stream>>>(y, Eemb, xin1, t);

        // layer 1: gates = [emb|ctx|h1] @ [Wih1|Whh1]^T + b
        gemm_bf16_wmma<0, 4, 1024><<<gLSTM, 128, 0, stream>>>(xin1, 1024, W1sw, b1s,
                                                              gates, nullptr, 2048, 2048, 2048);
        lstm_pw_kernel<<<256, 256, 0, stream>>>(gates, c1, xin2, 1024, xin1 + 512, 1024);

        // layer 2
        gemm_bf16_wmma<0, 4, 1024><<<gLSTM, 128, 0, stream>>>(xin2, 1024, W2sw, b2s,
                                                              gates, nullptr, 2048, 2048, 2048);
        lstm_pw_kernel<<<256, 256, 0, stream>>>(gates, c2, xin3, 1024, xin2 + 512, 1024);

        // layer 3 (h3 -> xin3[:,512:] for self-recurrence, and cdn[:,0:512])
        gemm_bf16_wmma<0, 4, 1024><<<gLSTM, 128, 0, stream>>>(xin3, 1024, W3sw, b3s,
                                                              gates, nullptr, 2048, 2048, 2048);
        lstm_pw_kernel<<<256, 256, 0, stream>>>(gates, c3, xin3 + 512, 1024, cdn, 768);

        // q = h3 @ Wq^T + bq   (A = xin3[:,512:1024])
        gemm_bf16_wmma<0, 4, 512><<<g256, 128, 0, stream>>>(xin3 + 512, 1024, Wqsw, bqp,
                                                            qbuf, nullptr, 256, 256, 256);

        // attention: softmax(q·keys) -> weights out, ctx -> xin1[:,256:512], cdn[:,512:768]
        attn_kernel<<<Bb, 256, 0, stream>>>(qbuf, keysb, valsb, attn_out, t, xin1, cdn);

        // s = tanh(cdn @ Wc^T + bc)  (bf16)
        gemm_bf16_wmma<1, 4, 768><<<g256, 128, 0, stream>>>(cdn, 768, Wcsw, bcp,
                                                            nullptr, sbuf, 256, 256, 256);

        // logits = s @ E^T + bprob  -> raw_out[b, t, 0:31]
        gemm_bf16_wmma<0, 1, 256><<<gV, 128, 0, stream>>>(sbuf, 256, Epsw, bpp,
                                                          raw_out + (size_t)t * Vv, nullptr,
                                                          Tt * Vv, 32, Vv);
    }
}
